// LSTM_20615843021484
// MI455X (gfx1250) — matrix operand order repaired and checked by reading, not compiled, not run on hardware
//
#include <hip/hip_runtime.h>

#define BATCH 256
#define SEQ   512
#define DIN   128
#define DHID  128
#define DOUT  128
#define BTILE 16
#define HS    136   // padded LDS row stride (halves) for the 128-wide h buffer

typedef _Float16 v16h __attribute__((ext_vector_type(16)));
typedef _Float16 h8   __attribute__((ext_vector_type(8)));
typedef float    v8f  __attribute__((ext_vector_type(8)));
typedef float    f4   __attribute__((ext_vector_type(4)));

__device__ __forceinline__ float fast_rcp(float x) {
  return __builtin_amdgcn_rcpf(x);
}
__device__ __forceinline__ float sigf(float x) {
  return fast_rcp(1.0f + __expf(-x));
}
__device__ __forceinline__ float tanh_fast(float x) {
  float e = __expf(-2.0f * x);
  return (1.0f - e) * fast_rcp(1.0f + e);
}

// Build a 16x32 f16 A-fragment row slice from LDS:
// halves 0..7  = base[0..7]   (K = k0 + hi*8 + j)
// halves 8..15 = base[16..23] (K = k0 + 16 + hi*8 + j)
__device__ __forceinline__ v16h load_a_lds(const _Float16* base) {
  h8 lo = *(const h8*)(base);
  h8 hh = *(const h8*)(base + 16);
  return __builtin_shufflevector(lo, hh, 0, 1, 2, 3, 4, 5, 6, 7,
                                 8, 9, 10, 11, 12, 13, 14, 15);
}

// B fragment (32x16, K x N) from a row-major [128][128] f32 global matrix:
// lane holds col n=lane%16, K = kc*32+hi*8+j (j<8) and +16 (j>=8), f32 -> f16.
__device__ __forceinline__ v16h load_b_global(const float* p) {
  v16h b;
#pragma unroll
  for (int j = 0; j < 16; ++j) b[j] = (_Float16)p[(j < 8 ? j : j + 8) * 128];
  return b;
}

#define WMMA(A, B, C) \
  __builtin_amdgcn_wmma_f32_16x16x32_f16(false, (A), false, (B), (short)0, (C), false, false)

// Issue raw f32 loads of one timestep's x row slice (A-fragment source data).
// raw[4*kc+0..1] = 8 floats at col kc*32+hi*8 ; raw[4*kc+2..3] = 8 floats at +16.
__device__ __forceinline__ void issue_x_loads(const float* xt, int hi, f4* raw) {
#pragma unroll
  for (int kc = 0; kc < 4; ++kc) {
    const float* p = xt + kc * 32 + hi * 8;
    raw[4 * kc + 0] = *(const f4*)(p + 0);
    raw[4 * kc + 1] = *(const f4*)(p + 4);
    raw[4 * kc + 2] = *(const f4*)(p + 16);
    raw[4 * kc + 3] = *(const f4*)(p + 20);
  }
}

__device__ __forceinline__ void cvt_x_frags(const f4* raw, v16h* Ax) {
#pragma unroll
  for (int kc = 0; kc < 4; ++kc) {
#pragma unroll
    for (int j = 0; j < 8; ++j)
      Ax[kc][j] = (_Float16)raw[4 * kc + (j >> 2)][j & 3];
#pragma unroll
    for (int j = 0; j < 8; ++j)
      Ax[kc][8 + j] = (_Float16)raw[4 * kc + 2 + (j >> 2)][j & 3];
  }
}

__global__ __launch_bounds__(256) void lstm_fused(
    const float* __restrict__ x,
    const float* __restrict__ x2i, const float* __restrict__ x2f,
    const float* __restrict__ x2g, const float* __restrict__ x2o,
    const float* __restrict__ h2i, const float* __restrict__ h2f,
    const float* __restrict__ h2g, const float* __restrict__ h2o,
    const float* __restrict__ w_out,
    float* __restrict__ out)
{
  // Double-buffered h tile: step t reads hL[t&1], writes h(t) to hL[(t+1)&1].
  // Readers and writers never share a buffer -> ONE barrier per step.
  __shared__ _Float16 hL[2][16 * HS];

  const int tid  = threadIdx.x;
  const int lane = tid & 31;
  const int wave = tid >> 5;          // 8 waves: wave owns hid cols [wave*16, +16)
  const int r    = lane & 15;
  const int hi   = lane >> 4;
  const int b0   = blockIdx.x * BTILE;
  const int gcol = wave * 16;

  // ---- zero the t=0 read buffer (h(-1) == 0) ----
  for (int idx = tid; idx < 16 * HS; idx += 256) hL[0][idx] = (_Float16)0.0f;

  // ---- BOTH weight matrices as register-resident B fragments (f16),
  //      loaded straight from global. Lane holds col gcol+r,
  //      rows kc*32+hi*8+{0..7,16..23} of each gate's 128x128 block.
  v16h whB[4][4];
  v16h wxB[4][4];
#pragma unroll
  for (int g = 0; g < 4; ++g) {
    const float* hp = (g == 0) ? h2i : (g == 1) ? h2f : (g == 2) ? h2g : h2o;
    const float* xp = (g == 0) ? x2i : (g == 1) ? x2f : (g == 2) ? x2g : x2o;
#pragma unroll
    for (int kc = 0; kc < 4; ++kc) {
      const size_t roff = (size_t)(kc * 32 + hi * 8) * DHID + gcol + r;
      whB[kc][g] = load_b_global(hp + roff);
      wxB[kc][g] = load_b_global(xp + roff);
    }
  }

  v8f cst = {0.f, 0.f, 0.f, 0.f, 0.f, 0.f, 0.f, 0.f};   // cell state, f32, persistent
  const float* xrow = x + (size_t)(b0 + r) * SEQ * DIN;

  // ---- prologue: stage x fragments for t = 0 ----
  v16h Ax[4];
  {
    f4 raw[16];
    issue_x_loads(xrow, hi, raw);
    cvt_x_frags(raw, Ax);
  }

  __syncthreads();                    // hL[0] zeroed and visible

  for (int t = 0; t < SEQ; ++t) {
    const _Float16* hRead  = hL[t & 1];
    _Float16*       hWrite = hL[(t + 1) & 1];

    // Issue next step's x loads immediately (waited on only at cvt below) and
    // prefetch t+4 into the cache hierarchy (global_prefetch).
    f4 raw[16];
    {
      const int tn = (t + 1 < SEQ) ? t + 1 : SEQ - 1;
      issue_x_loads(xrow + (size_t)tn * DIN, hi, raw);
      const int tp = (t + 4 < SEQ) ? t + 4 : SEQ - 1;
      __builtin_prefetch(xrow + (size_t)tp * DIN + hi * 64, 0, 3);
    }

    // h A-fragment LDS loads: latency overlaps the register-only x-WMMA block.
    v16h Ah[4];
#pragma unroll
    for (int kc = 0; kc < 4; ++kc)
      Ah[kc] = load_a_lds(hRead + r * HS + kc * 32 + hi * 8);

    v8f a0 = {0.f, 0.f, 0.f, 0.f, 0.f, 0.f, 0.f, 0.f};  // i gate pre-act
    v8f a1 = a0, a2 = a0, a3 = a0;                       // f, g, o

    // ---- x_t @ wx : A and B both register-resident, zero memory waits ----
#pragma unroll
    for (int kc = 0; kc < 4; ++kc) {
      a0 = WMMA(Ax[kc], wxB[kc][0], a0);
      a1 = WMMA(Ax[kc], wxB[kc][1], a1);
      a2 = WMMA(Ax[kc], wxB[kc][2], a2);
      a3 = WMMA(Ax[kc], wxB[kc][3], a3);
    }

    // ---- h_{t-1} @ wh : B register-resident; waits only on the Ah ds loads ----
#pragma unroll
    for (int kc = 0; kc < 4; ++kc) {
      a0 = WMMA(Ah[kc], whB[kc][0], a0);
      a1 = WMMA(Ah[kc], whB[kc][1], a1);
      a2 = WMMA(Ah[kc], whB[kc][2], a2);
      a3 = WMMA(Ah[kc], whB[kc][3], a3);
    }

    // ---- gates + state update (C layout: elem v -> row hi*8+v, col gcol+r) ----
    float hv[8];
#pragma unroll
    for (int v = 0; v < 8; ++v) {
      float iv = sigf(a0[v]);
      float fv = sigf(a1[v]);
      float gv = tanh_fast(a2[v]);
      float ov = sigf(a3[v]);
      float cn = fv * cst[v] + iv * gv;
      cst[v]   = cn;
      hv[v]    = ov * tanh_fast(cn);
    }

    // Convert next step's x (forces the load wait here, off the h critical path)
    cvt_x_frags(raw, Ax);

    // Publish h(t) into the *other* buffer; single barrier closes the step:
    // it publishes these writes AND fences this step's reads of hRead against
    // next step's writes (which target hRead's buffer).
#pragma unroll
    for (int v = 0; v < 8; ++v)
      hWrite[(hi * 8 + v) * HS + gcol + r] = (_Float16)hv[v];

    __syncthreads();
  }

  // Last write went to hL[SEQ & 1] == hL[0]; loop's final barrier covers it.
  const _Float16* hFin = hL[SEQ & 1];

  // ---- out = h_T @ w_out : wave owns out cols [gcol, gcol+16) ----
  v8f ao = {0.f, 0.f, 0.f, 0.f, 0.f, 0.f, 0.f, 0.f};
#pragma unroll
  for (int kc = 0; kc < 4; ++kc) {
    v16h Ah = load_a_lds(hFin + r * HS + kc * 32 + hi * 8);
    v16h Bo = load_b_global(w_out + (size_t)(kc * 32 + hi * 8) * DOUT + gcol + r);
    ao = WMMA(Ah, Bo, ao);
  }
#pragma unroll
  for (int v = 0; v < 8; ++v)
    out[(size_t)(b0 + hi * 8 + v) * DOUT + gcol + r] = ao[v];
}

extern "C" void kernel_launch(void* const* d_in, const int* in_sizes, int n_in,
                              void* d_out, int out_size, void* d_ws, size_t ws_size,
                              hipStream_t stream) {
  (void)in_sizes; (void)n_in; (void)d_ws; (void)ws_size; (void)out_size;
  const float* x    = (const float*)d_in[0];
  const float* x2i  = (const float*)d_in[1];
  const float* x2f  = (const float*)d_in[2];
  const float* x2g  = (const float*)d_in[3];
  const float* x2o  = (const float*)d_in[4];
  const float* h2i  = (const float*)d_in[5];
  const float* h2f  = (const float*)d_in[6];
  const float* h2g  = (const float*)d_in[7];
  const float* h2o  = (const float*)d_in[8];
  const float* wout = (const float*)d_in[9];
  float* out = (float*)d_out;

  lstm_fused<<<dim3(BATCH / BTILE), dim3(256), 0, stream>>>(
      x, x2i, x2f, x2g, x2o, h2i, h2f, h2g, h2o, wout, out);
}
